// MHA_30374008717431
// MI455X (gfx1250) — matrix operand order repaired and checked by reading, not compiled, not run on hardware
//
#include <hip/hip_runtime.h>
#include <hip/hip_bf16.h>

typedef __attribute__((ext_vector_type(16))) _Float16 v16h;
typedef __attribute__((ext_vector_type(8)))  _Float16 v8h;
typedef __attribute__((ext_vector_type(8)))  float    v8f;
typedef __attribute__((ext_vector_type(8)))  float    v8F;

#define BATCH   4
#define NSEQ    2048
#define DMODEL  512
#define HEADS   8
#define DHEAD   64
#define NQKV    1536   // 3*HEADS*DHEAD
#define MROWS   (BATCH * NSEQ)   // 8192

#define LOG2E   1.4426950408889634f
#define SM_SCALE 0.125f   // 1/sqrt(64)

static __device__ __forceinline__ v8f wmma_f16(v16h a, v16h b, v8f c) {
    return __builtin_amdgcn_wmma_f32_16x16x32_f16(false, a, false, b, (short)0, c,
                                                  false, false);
}

// ---------------------------------------------------------------------------
// Kernel 0a: convert x (f32) -> xh (f16), straight copy. 8 elems/thread.
// ---------------------------------------------------------------------------
__global__ __launch_bounds__(256) void
cvt_x_kernel(const float* __restrict__ in, _Float16* __restrict__ outp) {
    const int i = (blockIdx.x * 256 + threadIdx.x) * 8;
    const v8F v = *(const v8F*)(in + i);
    v8h o;
#pragma unroll
    for (int j = 0; j < 8; ++j) o[j] = (_Float16)v[j];
    *(v8h*)(outp + i) = o;
}

// ---------------------------------------------------------------------------
// Kernel 0b: W [512][1536] f32 -> W^T [1536][512] f16 (one-shot, tiny).
// ---------------------------------------------------------------------------
__global__ __launch_bounds__(256) void
cvt_wT_kernel(const float* __restrict__ w, _Float16* __restrict__ wT) {
    const int t  = blockIdx.x * 256 + threadIdx.x;   // 0 .. 98303
    const int n  = t >> 6;                            // output row (0..1535)
    const int k0 = (t & 63) * 8;                      // output col chunk
    v8h o;
#pragma unroll
    for (int j = 0; j < 8; ++j) o[j] = (_Float16)w[(size_t)(k0 + j) * NQKV + n];
    *(v8h*)(wT + (size_t)n * DMODEL + k0) = o;
}

// ---------------------------------------------------------------------------
// Kernel 1: qkv = xh @ W^T', scattered into f16 Q[bh][n][d], K[bh][n][d],
//           and transposed V^T[bh][d][n].
// Wave tile: 32 (M) x 64 (N).  Block: 4 waves -> 32 x 256.  8 WMMA / K-iter.
// The 64-wide, 64-aligned wave tile lies in exactly one of {Q,K,V} and one
// head, so section/head dispatch is wave-uniform (single branch).
// ---------------------------------------------------------------------------
__global__ __launch_bounds__(128) void
qkv_proj_kernel(const _Float16* __restrict__ xh, const _Float16* __restrict__ wT,
                _Float16* __restrict__ qb, _Float16* __restrict__ kb,
                _Float16* __restrict__ vtb) {
    const int lane = threadIdx.x & 31;
    const int wave = threadIdx.x >> 5;
    const int lm   = lane & 15;
    const int lh   = lane >> 4;
    const int koff = lh * 16;

    const int row0 = blockIdx.x * 32;                 // 0..8160
    const int col0 = blockIdx.y * 256 + wave * 64;    // 0..1535, 64-aligned

    v8f acc[2][4] = {};

    const _Float16* xrow0 = xh + (size_t)(row0 + lm) * DMODEL;
    const _Float16* xrow1 = xh + (size_t)(row0 + 16 + lm) * DMODEL;

    for (int kbase = 0; kbase < DMODEL; kbase += 32) {
        v16h a[2];
        {
            const v8h l0 = *(const v8h*)(xrow0 + kbase + koff);
            const v8h h0 = *(const v8h*)(xrow0 + kbase + 8 + koff);
            const v8h l1 = *(const v8h*)(xrow1 + kbase + koff);
            const v8h h1 = *(const v8h*)(xrow1 + kbase + 8 + koff);
#pragma unroll
            for (int i = 0; i < 8; ++i) {
                a[0][i] = l0[i]; a[0][8 + i] = h0[i];
                a[1][i] = l1[i]; a[1][8 + i] = h1[i];
            }
        }
#pragma unroll
        for (int ct = 0; ct < 4; ++ct) {
            const _Float16* wrow =
                wT + (size_t)(col0 + ct * 16 + lm) * DMODEL + kbase + lh * 16;
            const v16h b = *(const v16h*)wrow;
#pragma unroll
            for (int mt = 0; mt < 2; ++mt)
                acc[mt][ct] = wmma_f16(a[mt], b, acc[mt][ct]);
        }
    }

    // ---- epilogue: wave-uniform section/head dispatch ----
    const int section = col0 >> 9;          // 0=Q, 1=K, 2=V
    const int hh      = (col0 >> 6) & 7;    // head
    // Tile rows: m = row0 + mt*16 + r + lh*8.  row0 is 32-aligned so the whole
    // tile is in one batch image: bidx = row0>>11, nbase = row0&2047.
    const int bidx  = row0 >> 11;
    const int nbase = row0 & 2047;
    const size_t bhbase = ((size_t)bidx * HEADS + hh);

    if (section < 2) {
        // Q or K: [bh][n][64], element (n = nbase+mt*16+r+lh*8, d = ct*16+lm)
        _Float16* dst = (section == 0 ? qb : kb) + (bhbase * NSEQ + nbase) * DHEAD;
#pragma unroll
        for (int mt = 0; mt < 2; ++mt)
#pragma unroll
            for (int ct = 0; ct < 4; ++ct)
#pragma unroll
                for (int r = 0; r < 8; ++r)
                    dst[(size_t)(mt * 16 + r + lh * 8) * DHEAD + ct * 16 + lm] =
                        (_Float16)acc[mt][ct][r];
    } else {
        // V^T: [bh][d][n], element (d = ct*16+lm, n = nbase+mt*16+r+lh*8).
        // The 8 accumulator rows are consecutive n -> one b128 store per tile.
        _Float16* dst = vtb + bhbase * DHEAD * NSEQ + nbase;
#pragma unroll
        for (int mt = 0; mt < 2; ++mt)
#pragma unroll
            for (int ct = 0; ct < 4; ++ct) {
                v8h pk;
#pragma unroll
                for (int r = 0; r < 8; ++r) pk[r] = (_Float16)acc[mt][ct][r];
                *(v8h*)(dst + (size_t)(ct * 16 + lm) * NSEQ + mt * 16 + lh * 8) = pk;
            }
    }
}

// ---------------------------------------------------------------------------
// Kernel 2: flash attention.  grid = (n/64, b*h), block = 128 (4 waves).
// Each wave: 16 query rows, full d=64 output, online softmax over 32-key tiles.
// ---------------------------------------------------------------------------
__global__ __launch_bounds__(128) void
flash_attn_kernel(const _Float16* __restrict__ q, const _Float16* __restrict__ k,
                  const _Float16* __restrict__ vt, float* __restrict__ out) {
    __shared__ __align__(16) _Float16 pbuf[4][16 * 32];

    const int lane = threadIdx.x & 31;
    const int wave = threadIdx.x >> 5;
    const int lm   = lane & 15;
    const int lh   = lane >> 4;
    const int koff = lh * 16;

    const int bh = blockIdx.y;          // 0..31
    const int bb = bh >> 3;
    const int hh = bh & 7;
    const int q0 = blockIdx.x * 64 + wave * 16;

    const _Float16* qp = q  + (size_t)bh * NSEQ * DHEAD;
    const _Float16* kp = k  + (size_t)bh * NSEQ * DHEAD;
    const _Float16* vp = vt + (size_t)bh * DHEAD * NSEQ;

    // Q fragments for d-chunks [0,32) and [32,64)
    v16h qf[2];
    {
        const _Float16* qrow = qp + (size_t)(q0 + lm) * DHEAD;
#pragma unroll
        for (int dk = 0; dk < 2; ++dk) {
            const v8h lo = *(const v8h*)(qrow + dk * 32 + koff);
            const v8h hi = *(const v8h*)(qrow + dk * 32 + 8 + koff);
#pragma unroll
            for (int i = 0; i < 8; ++i) { qf[dk][i] = lo[i]; qf[dk][8 + i] = hi[i]; }
        }
    }

    v8f   acc[4] = {};
    float mrow[8], lrow[8];
#pragma unroll
    for (int r = 0; r < 8; ++r) { mrow[r] = -INFINITY; lrow[r] = 0.0f; }

    _Float16* myp = pbuf[wave];

    for (int j0 = 0; j0 < NSEQ; j0 += 32) {
        // ---- S = Q * K^T  (16 x 32 tile of scores) ----
        v8f s[2] = {};
#pragma unroll
        for (int jt = 0; jt < 2; ++jt) {
#pragma unroll
            for (int dk = 0; dk < 2; ++dk) {
                const _Float16* krow =
                    kp + (size_t)(j0 + jt * 16 + lm) * DHEAD + dk * 32 + lh * 16;
                const v16h bf = *(const v16h*)krow;
                s[jt] = wmma_f16(qf[dk], bf, s[jt]);
            }
        }

        // ---- online softmax ----
        float corr[8];
#pragma unroll
        for (int r = 0; r < 8; ++r) {
            float s0 = s[0][r] * SM_SCALE;
            float s1 = s[1][r] * SM_SCALE;
            float t = fmaxf(s0, s1);
            t = fmaxf(t, __shfl_xor(t, 1, 32));
            t = fmaxf(t, __shfl_xor(t, 2, 32));
            t = fmaxf(t, __shfl_xor(t, 4, 32));
            t = fmaxf(t, __shfl_xor(t, 8, 32));
            const float mn = fmaxf(mrow[r], t);
            corr[r] = exp2f((mrow[r] - mn) * LOG2E);
            const float p0 = exp2f((s0 - mn) * LOG2E);
            const float p1 = exp2f((s1 - mn) * LOG2E);
            s[0][r] = p0;
            s[1][r] = p1;
            float ps = p0 + p1;
            ps += __shfl_xor(ps, 1, 32);
            ps += __shfl_xor(ps, 2, 32);
            ps += __shfl_xor(ps, 4, 32);
            ps += __shfl_xor(ps, 8, 32);
            lrow[r] = lrow[r] * corr[r] + ps;
            mrow[r] = mn;
        }
#pragma unroll
        for (int dt = 0; dt < 4; ++dt)
#pragma unroll
            for (int r = 0; r < 8; ++r)
                acc[dt][r] *= corr[r];

        // ---- bounce P through LDS: C-layout -> A-fragment layout ----
#pragma unroll
        for (int jt = 0; jt < 2; ++jt)
#pragma unroll
            for (int r = 0; r < 8; ++r)
                myp[(r + lh * 8) * 32 + jt * 16 + lm] = (_Float16)s[jt][r];
        asm volatile("s_wait_dscnt 0" ::: "memory");

        v16h pf;
        {
            const _Float16* pr = myp + lm * 32;
            const v8h lo = *(const v8h*)(pr + koff);
            const v8h hi = *(const v8h*)(pr + 8 + koff);
#pragma unroll
            for (int i = 0; i < 8; ++i) { pf[i] = lo[i]; pf[8 + i] = hi[i]; }
        }

        // ---- acc += P * V  (V^T rows are contiguous in K=j) ----
#pragma unroll
        for (int dt = 0; dt < 4; ++dt) {
            const _Float16* vrow =
                vp + (size_t)(dt * 16 + lm) * NSEQ + j0 + lh * 16;
            const v16h bf = *(const v16h*)vrow;
            acc[dt] = wmma_f16(pf, bf, acc[dt]);
        }
    }

    // ---- epilogue: divide by row sum, store [b, n, h*d] f32 ----
#pragma unroll
    for (int dt = 0; dt < 4; ++dt)
#pragma unroll
        for (int r = 0; r < 8; ++r) {
            const int nrow = q0 + r + lh * 8;
            out[((size_t)bb * NSEQ + nrow) * DMODEL + hh * DHEAD + dt * 16 + lm] =
                acc[dt][r] / lrow[r];
        }
}

// ---------------------------------------------------------------------------
extern "C" void kernel_launch(void* const* d_in, const int* in_sizes, int n_in,
                              void* d_out, int out_size, void* d_ws, size_t ws_size,
                              hipStream_t stream) {
    const float* x = (const float*)d_in[0];
    const float* w = (const float*)d_in[1];
    float* out = (float*)d_out;

    const size_t QELEMS = (size_t)BATCH * HEADS * NSEQ * DHEAD; // 4,194,304
    _Float16* qb  = (_Float16*)d_ws;
    _Float16* kb  = qb + QELEMS;
    _Float16* vtb = kb + QELEMS;
    _Float16* xh  = vtb + QELEMS;                       // [8192][512] f16
    _Float16* wT  = xh + (size_t)MROWS * DMODEL;        // [1536][512] f16

    // 0a: x f32 -> f16   (8192*512 / (256*8) = 2048 blocks)
    cvt_x_kernel<<<dim3(2048), dim3(256), 0, stream>>>(x, xh);
    // 0b: W f32 -> W^T f16  (1536*512 / (256*8) = 384 blocks)
    cvt_wT_kernel<<<dim3(384), dim3(256), 0, stream>>>(w, wT);

    // QKV projection: grid (8192/32, 1536/256)
    qkv_proj_kernel<<<dim3(256, 6), dim3(128), 0, stream>>>(xh, wT, qb, kb, vtb);

    // Flash attention: grid (2048/64, b*h)
    flash_attn_kernel<<<dim3(32, 32), dim3(128), 0, stream>>>(qb, kb, vtb, out);
}
